// SparseBlock_75892072120727
// MI455X (gfx1250) — compile-verified, matrix operand order checked by reading, not run on hardware
//
#include <hip/hip_runtime.h>

// ---------------------------------------------------------------------------
// Sparse block conv (SBNet-style) for MI455X / gfx1250, wave32 + WMMA.
// GEMM view per active block: M=1024 px, N=16 c_out, K=144 (pad 160).
// v_wmma_f32_16x16x32_f16, fp32 accumulate. Bandwidth-bound problem:
// only inactive output blocks are zeroed (mask in d_ws), active blocks are
// written exactly once by the conv kernel. Two independent output tiles are
// processed per wave iteration so the two WMMA accumulation chains overlap.
// ---------------------------------------------------------------------------

typedef __attribute__((ext_vector_type(16))) _Float16 v16h;
typedef __attribute__((ext_vector_type(8)))  _Float16 half8;
typedef __attribute__((ext_vector_type(4)))  _Float16 half4;
typedef __attribute__((ext_vector_type(8)))  float    v8f;
typedef __attribute__((ext_vector_type(4)))  float    f4;

#define HW    1536
#define CIN   16
#define COUT  16
#define TS    34      // gathered input tile side (32 + 2 halo)
#define BS    32      // output block side
#define NBLK  48      // blocks per image side
#define NACT  1152    // active blocks

// --- kernel 1: build 48x48 activity mask in workspace -----------------------
__global__ void build_mask_kernel(const int* __restrict__ idx,
                                  unsigned char* __restrict__ mask) {
    for (int i = threadIdx.x; i < NBLK * NBLK; i += blockDim.x) mask[i] = 0;
    __syncthreads();
    for (int i = threadIdx.x; i < NACT; i += blockDim.x) {
        int bi = idx[i * 3 + 1];
        int bj = idx[i * 3 + 2];
        mask[bi * NBLK + bj] = 1;
    }
}

// --- kernel 2: zero only the inactive 32x32x16 output blocks ----------------
__global__ void zero_inactive_kernel(const unsigned char* __restrict__ mask,
                                     float* __restrict__ out) {
    int b = blockIdx.x;                 // 0 .. 2303
    if (mask[b]) return;
    int bi = b / NBLK, bj = b % NBLK;
    const f4 z = {0.f, 0.f, 0.f, 0.f};
    for (int e = threadIdx.x; e < BS * BS * COUT / 4; e += blockDim.x) {
        int q   = e & 3;                // which float4 of the 16 channels
        int pix = e >> 2;
        int py  = pix >> 5, px = pix & 31;
        size_t off = ((size_t)(bi * BS + py) * HW + (bj * BS + px)) * COUT + q * 4;
        *(f4*)(out + off) = z;
    }
}

// Load one A/B fragment half-pair from LDS and splice into a v16h.
// Per ISA 16-bit layout: frag elems 0..7 = ci {laneHi*8..+7} of group g0,
// elems 8..15 = same ci range of group g1 -> two ds_load_b128.
__device__ __forceinline__ v16h make_frag(const _Float16* p0, const _Float16* p1) {
    half8 lo = *(const half8*)p0;
    half8 hi = *(const half8*)p1;
    return __builtin_shufflevector(lo, hi, 0, 1, 2, 3, 4, 5, 6, 7,
                                            8, 9, 10, 11, 12, 13, 14, 15);
}

// --- kernel 3: gather + implicit-GEMM conv via WMMA + scatter ---------------
__global__ void __launch_bounds__(256)
sparse_conv_wmma(const float* __restrict__ inp, const float* __restrict__ cw,
                 const float* __restrict__ cb, const int* __restrict__ idx,
                 float* __restrict__ out) {
    // f16 input tile: 34*34*16 halves = 36,992 B
    __shared__ __align__(16) _Float16 sIn[TS * TS * CIN];
    // transposed f16 weights wT[g][n][ci], g = ky*3+kx; group 9 is the zero K-pad;
    // the A-side pad reads zeros from sZero.
    __shared__ __align__(16) _Float16 sWT[10 * 16 * 16];
    // 16 zero halves for the A fragment's padded half in chunk 4
    __shared__ __align__(16) _Float16 sZero[16];

    const int tid = threadIdx.x;
    const int blk = blockIdx.x;
    const int bi = idx[blk * 3 + 1];
    const int bj = idx[blk * 3 + 2];
    const int row0 = bi * BS - 1;       // halo origin in unpadded image
    const int col0 = bj * BS - 1;

    // ---- stage input tile: fp32 global -> f16 LDS (vector loads/stores) ----
    for (int e = tid; e < TS * TS * 4; e += 256) {
        int c4  = e & 3;                // group of 4 channels
        int pix = e >> 2;
        int y = pix / TS, x = pix % TS;
        int r = row0 + y, c = col0 + x;
        f4 v = {0.f, 0.f, 0.f, 0.f};
        if ((unsigned)r < HW && (unsigned)c < HW)
            v = *(const f4*)(inp + ((size_t)r * HW + c) * CIN + c4 * 4);
        half4 h;
        h.x = (_Float16)v.x; h.y = (_Float16)v.y;
        h.z = (_Float16)v.z; h.w = (_Float16)v.w;
        *(half4*)&sIn[pix * CIN + c4 * 4] = h;   // 8B-aligned ds store
    }

    // ---- stage weights transposed: cw[ky][kx][ci][n] -> wT[g][n][ci] -------
    for (int e = tid; e < 9 * 16 * 16; e += 256) {
        int n  = e & 15;
        int ci = (e >> 4) & 15;
        int g  = e >> 8;
        sWT[(g * 16 + n) * 16 + ci] = (_Float16)cw[e];
    }
    sWT[9 * 256 + tid] = (_Float16)0.f;          // zero K-pad group (256 thr)
    if (tid < 16) sZero[tid] = (_Float16)0.f;
    __syncthreads();

    const int lane   = tid & 31;
    const int laneLo = lane & 15;
    const int laneHi = lane >> 4;
    const int wave   = tid >> 5;                  // 8 waves
    const float bn   = cb[laneLo];                // bias for this lane's N

    // ---- B fragments for the 5 K-chunks (chunk 4 upper half = zeros) -------
    v16h bfrag[5];
    {
        const _Float16* pB = &sWT[laneLo * 16 + laneHi * 8];
#pragma unroll
        for (int c = 0; c < 5; ++c)
            bfrag[c] = make_frag(pB + (2 * c) * 256, pB + (2 * c + 1) * 256);
    }

    // ---- 64 tiles of 16 px; each wave handles two independent tiles/iter ---
    const _Float16* pZ = &sZero[laneHi * 8];
    for (int it = 0; it < 4; ++it) {
        const int t0 = it * 16 + wave;           // tiles t0 and t0+8
        const int py0 = t0 >> 1,      px00 = (t0 & 1) << 4;
        const int t1  = t0 + 8;
        const int py1 = t1 >> 1,      px01 = (t1 & 1) << 4;

        // one LDS base per tile; (ky*34+kx)*16 folds into ds-load offsets
        const _Float16* pA0 = &sIn[(py0 * TS + px00 + laneLo) * CIN + laneHi * 8];
        const _Float16* pA1 = &sIn[(py1 * TS + px01 + laneLo) * CIN + laneHi * 8];

        v8f acc0 = {0.f, 0.f, 0.f, 0.f, 0.f, 0.f, 0.f, 0.f};
        v8f acc1 = acc0;
#pragma unroll
        for (int c = 0; c < 5; ++c) {
            const int g0 = 2 * c, g1 = 2 * c + 1;
            const int o0 = ((g0 / 3) * TS + (g0 % 3)) * CIN;   // constexpr
            const int o1 = (g1 < 9) ? ((g1 / 3) * TS + (g1 % 3)) * CIN : 0;

            v16h a0 = (g1 < 9) ? make_frag(pA0 + o0, pA0 + o1)
                               : make_frag(pA0 + o0, pZ);
            v16h a1 = (g1 < 9) ? make_frag(pA1 + o0, pA1 + o1)
                               : make_frag(pA1 + o0, pZ);

            acc0 = __builtin_amdgcn_wmma_f32_16x16x32_f16(
                false, a0, false, bfrag[c], (short)0, acc0, false, false);
            acc1 = __builtin_amdgcn_wmma_f32_16x16x32_f16(
                false, a1, false, bfrag[c], (short)0, acc1, false, false);
        }

        // D layout: lane holds N=laneLo, rows M = laneHi*8 + r (r=0..7).
        // Consecutive lanes -> consecutive channels -> 64B coalesced stores.
        float* op0 = out + ((size_t)(bi * BS + py0) * HW
                            + (bj * BS + px00 + laneHi * 8)) * COUT + laneLo;
        float* op1 = out + ((size_t)(bi * BS + py1) * HW
                            + (bj * BS + px01 + laneHi * 8)) * COUT + laneLo;
#pragma unroll
        for (int r = 0; r < 8; ++r) {
            op0[(size_t)r * COUT] = acc0[r] + bn;
            op1[(size_t)r * COUT] = acc1[r] + bn;
        }
    }
}

extern "C" void kernel_launch(void* const* d_in, const int* in_sizes, int n_in,
                              void* d_out, int out_size, void* d_ws, size_t ws_size,
                              hipStream_t stream) {
    const float* inp  = (const float*)d_in[0];
    const float* cw   = (const float*)d_in[1];
    const float* cb   = (const float*)d_in[2];
    const int*   aidx = (const int*)d_in[3];
    float* out = (float*)d_out;
    unsigned char* mask = (unsigned char*)d_ws;   // 2304 bytes

    build_mask_kernel<<<1, 256, 0, stream>>>(aidx, mask);
    zero_inactive_kernel<<<NBLK * NBLK, 256, 0, stream>>>(mask, out);
    sparse_conv_wmma<<<NACT, 256, 0, stream>>>(inp, cw, cb, aidx, out);
}
